// MultiHeadAttention_27384711479865
// MI455X (gfx1250) — matrix operand order repaired and checked by reading, not compiled, not run on hardware
//
#include <hip/hip_runtime.h>
#include <hip/hip_bf16.h>

// ---------- types ----------
typedef __attribute__((ext_vector_type(16))) __bf16 v16bf;
typedef __attribute__((ext_vector_type(8)))  float  v8f;

__device__ __forceinline__ unsigned short f2bf(float f) {
  unsigned u = __float_as_uint(f);
  return (unsigned short)((u + 0x7FFFu + ((u >> 16) & 1u)) >> 16);
}

union FragU { uint4 q[2]; v16bf bf; };

// ---------- CDNA5 async global->LDS (portable inline asm; ASYNCcnt tracked) ----------
__device__ __forceinline__ void async_load16(unsigned lds_off, const void* gaddr) {
  asm volatile("global_load_async_to_lds_b128 %0, %1, off"
               :
               : "v"(lds_off), "v"((unsigned long long)(size_t)gaddr)
               : "memory");
}
__device__ __forceinline__ void wait_async0() {
  asm volatile("s_wait_asynccnt 0x0" ::: "memory");
}

// ---------- fp32 -> bf16 conversion (vectorized x4) ----------
__global__ __launch_bounds__(256) void cvt_kernel(const float4* __restrict__ src,
                                                  ushort4* __restrict__ dst,
                                                  long long n4) {
  long long i = (long long)blockIdx.x * blockDim.x + threadIdx.x;
  if (i < n4) {
    float4 f = src[i];
    ushort4 o;
    o.x = f2bf(f.x); o.y = f2bf(f.y); o.z = f2bf(f.z); o.w = f2bf(f.w);
    dst[i] = o;
  }
}

// ---------- bf16 [b][S][D] -> [b][D][S] transpose ----------
__global__ __launch_bounds__(256) void transpose_kernel(const unsigned short* __restrict__ src,
                                                        unsigned short* __restrict__ dst,
                                                        int S, int D) {
  __shared__ unsigned short tile[32][33];
  const int b  = blockIdx.z;
  const int d0 = blockIdx.x * 32;
  const int s0 = blockIdx.y * 32;
  src += (long long)b * S * D;
  dst += (long long)b * S * D;
  const int tx = threadIdx.x;      // 0..31
  const int ty = threadIdx.y;      // 0..7
  for (int i = 0; i < 32; i += 8)
    tile[ty + i][tx] = src[(long long)(s0 + ty + i) * D + d0 + tx];
  __syncthreads();
  for (int i = 0; i < 32; i += 8)
    dst[(long long)(d0 + ty + i) * S + s0 + tx] = tile[tx][ty + i];
}

// ---------- row softmax: fp32 scores [rows][n] -> bf16 probs ----------
__global__ __launch_bounds__(256) void softmax_kernel(const float* __restrict__ scores,
                                                      unsigned short* __restrict__ attn,
                                                      int ncols) {
  __shared__ float red[256];
  const long long row = blockIdx.x;
  const float* srow = scores + row * (long long)ncols;
  unsigned short* arow = attn + row * (long long)ncols;
  const int tid = threadIdx.x;          // 256 threads * 8 = 2048 cols
  float vals[8];
  float m = -INFINITY;
  for (int i = 0; i < 8; ++i) {
    vals[i] = srow[tid * 8 + i];
    m = fmaxf(m, vals[i]);
  }
  red[tid] = m; __syncthreads();
  for (int s = 128; s > 0; s >>= 1) {
    if (tid < s) red[tid] = fmaxf(red[tid], red[tid + s]);
    __syncthreads();
  }
  m = red[0]; __syncthreads();
  float sum = 0.f;
  for (int i = 0; i < 8; ++i) {
    vals[i] = __expf(vals[i] - m);
    sum += vals[i];
  }
  red[tid] = sum; __syncthreads();
  for (int s = 128; s > 0; s >>= 1) {
    if (tid < s) red[tid] += red[tid + s];
    __syncthreads();
  }
  const float inv = 1.f / red[0];
  for (int i = 0; i < 8; ++i)
    arow[tid * 8 + i] = f2bf(vals[i] * inv);
}

// ---------- WMMA GEMM: C[M][N] = scale * (A[M][K] * Bsrc[N][K]^T) + bias[N] ----------
// A, Bsrc are bf16 row-major; C is fp32 or bf16. All dims multiples of tile sizes.
// Double-buffered: global->LDS via async-to-LDS instructions overlapped with WMMA.
#define BM 128
#define BN 128
#define BK 32
#define BUFB (BM * BK * 2)   // bytes per LDS buffer (8 KB)

__global__ __launch_bounds__(256) void wmma_gemm_kernel(
    const unsigned short* __restrict__ A, long long lda, long long aBatch,
    const unsigned short* __restrict__ Bsrc, long long ldb, long long bBatch,
    void* __restrict__ Cout, long long ldc, long long cBatch,
    int K, const float* __restrict__ bias, float scale, int outIsBf16) {
  __shared__ __align__(128) unsigned short lA[2][BM * BK];
  __shared__ __align__(128) unsigned short lB[2][BN * BK];

  const int zb = blockIdx.z;
  A    += (long long)zb * aBatch;
  Bsrc += (long long)zb * bBatch;

  const int tid  = threadIdx.x;
  const int m0   = blockIdx.y * BM;
  const int n0   = blockIdx.x * BN;
  const int wave = tid >> 5;
  const int lane = tid & 31;
  const int wr   = (wave & 1) * 64;   // 2 waves along M: 64 rows each
  const int wc   = (wave >> 1) * 32;  // 4 waves along N: 32 cols each
  const int half = lane >> 4;
  const int l16  = lane & 15;

  // cooperative tile-load mapping: 256 threads, 4x16B each per tile
  const int lrow = tid >> 2;          // 0..63
  const int lcol = (tid & 3) * 8;     // bf16 column within 32-wide tile

  // per-thread global source pointers (advance by BK per K-step)
  const unsigned short* gA0 = A    + (long long)(m0 + lrow)      * lda + lcol;
  const unsigned short* gA1 = A    + (long long)(m0 + lrow + 64) * lda + lcol;
  const unsigned short* gB0 = Bsrc + (long long)(n0 + lrow)      * ldb + lcol;
  const unsigned short* gB1 = Bsrc + (long long)(n0 + lrow + 64) * ldb + lcol;

  // per-thread LDS destination byte offsets (buffer 0)
  const unsigned aoff0 = (unsigned)(size_t)&lA[0][lrow * BK + lcol];
  const unsigned aoff1 = (unsigned)(size_t)&lA[0][(lrow + 64) * BK + lcol];
  const unsigned boff0 = (unsigned)(size_t)&lB[0][lrow * BK + lcol];
  const unsigned boff1 = (unsigned)(size_t)&lB[0][(lrow + 64) * BK + lcol];

  v8f acc[4][2];
  const v8f vzero = {0.f, 0.f, 0.f, 0.f, 0.f, 0.f, 0.f, 0.f};
  for (int mt = 0; mt < 4; ++mt)
    for (int nt = 0; nt < 2; ++nt)
      acc[mt][nt] = vzero;

  // prologue: tile 0 -> buffer 0
  async_load16(aoff0, gA0);
  async_load16(aoff1, gA1);
  async_load16(boff0, gB0);
  async_load16(boff1, gB1);

  const int ntiles = K / BK;
  for (int t = 0; t < ntiles; ++t) {
    wait_async0();          // our tile-t async writes have landed in LDS
    __syncthreads();        // everyone's writes visible; all tile-(t-1) reads done

    const int cur = t & 1;
    if (t + 1 < ntiles) {   // issue tile t+1 into the other buffer, overlap with compute
      const unsigned nb = (unsigned)((t + 1) & 1) * BUFB;
      gA0 += BK; gA1 += BK; gB0 += BK; gB1 += BK;
      async_load16(aoff0 + nb, gA0);
      async_load16(aoff1 + nb, gA1);
      async_load16(boff0 + nb, gB0);
      async_load16(boff1 + nb, gB1);
    }

    // A fragments (16x32 bf16): lane(half,m) holds K in [16h, 16h+16)
    v16bf afrag[4];
    for (int mt = 0; mt < 4; ++mt) {
      FragU fa;
      const unsigned short* p = &lA[cur][(wr + mt * 16 + l16) * BK + half * 16];
      fa.q[0] = *(const uint4*)p;
      fa.q[1] = *(const uint4*)(p + 8);
      afrag[mt] = fa.bf;
    }
    // B fragments (32x16 bf16): lane(half,n) holds column n, K in [16h, 16h+16)
    v16bf bfrag[2];
    for (int nt = 0; nt < 2; ++nt) {
      const unsigned short* p = &lB[cur][(wc + nt * 16 + l16) * BK + half * 16];
      bfrag[nt] = *(const v16bf*)p;   // 32B-aligned contiguous
    }

    for (int mt = 0; mt < 4; ++mt)
      for (int nt = 0; nt < 2; ++nt)
        acc[mt][nt] = __builtin_amdgcn_wmma_f32_16x16x32_bf16(
            false, afrag[mt], false, bfrag[nt], (short)0, acc[mt][nt],
            false, false);
  }

  // D layout: VGPR r -> row = base + r + 8*half, col = base + (lane&15)
  for (int mt = 0; mt < 4; ++mt)
    for (int nt = 0; nt < 2; ++nt) {
      const int rowb = m0 + wr + mt * 16 + half * 8;
      const int col  = n0 + wc + nt * 16 + l16;
      const float bv = bias ? bias[col] : 0.f;
      for (int r = 0; r < 8; ++r) {
        const float vv = acc[mt][nt][r] * scale + bv;
        const long long idx =
            (long long)zb * cBatch + (long long)(rowb + r) * ldc + col;
        if (outIsBf16) ((unsigned short*)Cout)[idx] = f2bf(vv);
        else           ((float*)Cout)[idx] = vv;
      }
    }
}

// ---------- host-side orchestration ----------
extern "C" void kernel_launch(void* const* d_in, const int* in_sizes, int n_in,
                              void* d_out, int out_size, void* d_ws, size_t ws_size,
                              hipStream_t stream) {
  (void)in_sizes; (void)n_in; (void)out_size; (void)ws_size;

  const float* x  = (const float*)d_in[0];
  const float* Wq = (const float*)d_in[1];
  const float* bq = (const float*)d_in[2];
  const float* Wk = (const float*)d_in[3];
  const float* bk = (const float*)d_in[4];
  const float* Wv = (const float*)d_in[5];
  const float* bv = (const float*)d_in[6];
  const float* Wo = (const float*)d_in[7];
  const float* bo = (const float*)d_in[8];

  const int B = 4, S = 2048, D = 1024;
  const long long nX  = (long long)B * S * D;   // 8,388,608
  const long long nW  = (long long)D * D;       // 1,048,576
  const long long nSc = (long long)B * S * S;   // 16,777,216

  char* p = (char*)d_ws;
  unsigned short* xb   = (unsigned short*)p; p += nX * 2;
  unsigned short* Wqb  = (unsigned short*)p; p += nW * 2;
  unsigned short* Wkb  = (unsigned short*)p; p += nW * 2;
  unsigned short* Wvb  = (unsigned short*)p; p += nW * 2;
  unsigned short* Wob  = (unsigned short*)p; p += nW * 2;
  unsigned short* q    = (unsigned short*)p; p += nX * 2;
  unsigned short* kmat = (unsigned short*)p; p += nX * 2;
  unsigned short* v    = (unsigned short*)p; p += nX * 2;
  unsigned short* vt   = (unsigned short*)p; p += nX * 2;
  float*          sc   = (float*)p;          p += nSc * 4;
  unsigned short* attn = (unsigned short*)p; p += nSc * 2;
  unsigned short* ctx  = (unsigned short*)p; p += nX * 2;

  // 1) fp32 -> bf16 conversions
  cvt_kernel<<<(int)(nX / 4 / 256), 256, 0, stream>>>((const float4*)x,  (ushort4*)xb,  nX / 4);
  cvt_kernel<<<(int)(nW / 4 / 256), 256, 0, stream>>>((const float4*)Wq, (ushort4*)Wqb, nW / 4);
  cvt_kernel<<<(int)(nW / 4 / 256), 256, 0, stream>>>((const float4*)Wk, (ushort4*)Wkb, nW / 4);
  cvt_kernel<<<(int)(nW / 4 / 256), 256, 0, stream>>>((const float4*)Wv, (ushort4*)Wvb, nW / 4);
  cvt_kernel<<<(int)(nW / 4 / 256), 256, 0, stream>>>((const float4*)Wo, (ushort4*)Wob, nW / 4);

  // 2) Q/K/V projections: [8192 x 1024] = xb * W^T + b
  dim3 gP(D / BN, (B * S) / BM, 1);   // (8, 64, 1)
  wmma_gemm_kernel<<<gP, 256, 0, stream>>>(xb, D, 0, Wqb, D, 0, q,    D, 0, D, bq, 1.f, 1);
  wmma_gemm_kernel<<<gP, 256, 0, stream>>>(xb, D, 0, Wkb, D, 0, kmat, D, 0, D, bk, 1.f, 1);
  wmma_gemm_kernel<<<gP, 256, 0, stream>>>(xb, D, 0, Wvb, D, 0, v,    D, 0, D, bv, 1.f, 1);

  // 3) v [b][S][D] -> vt [b][D][S]
  transpose_kernel<<<dim3(D / 32, S / 32, B), dim3(32, 8), 0, stream>>>(v, vt, S, D);

  // 4) scores = (q @ k^T) / sqrt(D), fp32, per batch
  dim3 gS(S / BN, S / BM, B);         // (16, 16, 4)
  wmma_gemm_kernel<<<gS, 256, 0, stream>>>(q, D, (long long)S * D,
                                           kmat, D, (long long)S * D,
                                           sc, S, (long long)S * S,
                                           D, nullptr, 0.03125f, 0);

  // 5) softmax rows -> bf16 probs
  softmax_kernel<<<B * S, 256, 0, stream>>>(sc, attn, S);

  // 6) ctx = attn @ v  (Bsrc = vt[b] is [D][S] row-major = v^T)
  dim3 gC(D / BN, S / BM, B);         // (8, 16, 4)
  wmma_gemm_kernel<<<gC, 256, 0, stream>>>(attn, S, (long long)S * S,
                                           vt, S, (long long)D * S,
                                           ctx, D, (long long)S * D,
                                           S, nullptr, 1.f, 1);

  // 7) out = ctx @ Wo^T + bo, fp32 to d_out
  wmma_gemm_kernel<<<gP, 256, 0, stream>>>(ctx, D, 0, Wob, D, 0,
                                           (float*)d_out, D, 0, D, bo, 1.f, 0);
}
